// TwoDGRUClassifier_77807627534337
// MI455X (gfx1250) — matrix operand order repaired and checked by reading, not compiled, not run on hardware
//
#include <hip/hip_runtime.h>
#include <hip/hip_bf16.h>

typedef __attribute__((ext_vector_type(16))) __bf16 v16bf;
typedef __attribute__((ext_vector_type(8)))  __bf16 v8bf;
typedef __attribute__((ext_vector_type(8)))  float  v8f;

#define HIDN 128
#define GRID 32
#define BQ   32
#define EMBD 64
#define NDIRS 4
#define NCLSS 1000
#define ASTRIDE 264   // 256 + 8 bf16 pad -> 528B row stride, conflict-free 16B ds loads

__device__ __forceinline__ float  bf2f(__bf16 v) { return (float)v; }
__device__ __forceinline__ __bf16 f2bf(float v)  { return (__bf16)v; }

// ---------------------------------------------------------------------------
// Prepack f32 (K x Nsrc, row-major) weight matrix into bf16 WMMA B-fragment
// layout: [K/32][NT][lane(32)][elem(16)].  Lane L covers column nt*16+(L&15);
// elements e<8 cover K = kt*32 + kg + e, e>=8 cover kt*32 + 16 + kg + (e-8),
// kg = (L>=16)?8:0.  Zero-pads N.
// ---------------------------------------------------------------------------
__global__ void prepack_kernel(const float* __restrict__ src, __bf16* __restrict__ dst,
                               int K, int Nsrc, int NT) {
    int total = (K >> 5) * NT * 512;
    int idx = blockIdx.x * 256 + threadIdx.x;
    if (idx >= total) return;
    int e  = idx & 15;
    int L  = (idx >> 4) & 31;
    int nt = (idx >> 9) % NT;
    int kt = idx / (NT * 512);
    int k  = kt * 32 + ((e < 8) ? (((L >= 16) ? 8 : 0) + e) : (16 + ((L >= 16) ? 8 : 0) + (e - 8)));
    int n  = nt * 16 + (L & 15);
    float v = (n < Nsrc) ? src[(size_t)k * Nsrc + n] : 0.0f;
    dst[idx] = f2bf(v);
}

// ---------------------------------------------------------------------------
// Embedding: emb[(i,j,b,e)] = be[e] + sum_c x[b,i,j,c] * We[c,e]   (bf16 out)
// ---------------------------------------------------------------------------
__global__ void embed_kernel(const float* __restrict__ x, const float* __restrict__ We,
                             const float* __restrict__ be, __bf16* __restrict__ emb) {
    int idx = blockIdx.x * 256 + threadIdx.x;          // 2^21 total
    int e = idx & 63;
    int b = (idx >> 6) & 31;
    int j = (idx >> 11) & 31;
    int i = (idx >> 16) & 31;
    const float* xp = x + (((size_t)b * GRID + i) * GRID + j) * 3;
    float v = be[e] + xp[0] * We[0 * EMBD + e] + xp[1] * We[1 * EMBD + e]
                    + xp[2] * We[2 * EMBD + e];
    emb[idx] = f2bf(v);
}

// ---------------------------------------------------------------------------
// One anti-diagonal wavefront step of the 2-D GRU scan.
// Block = 256 threads = 8 waves; one block per (direction, cell-on-diagonal).
// A = [x | 0.5*(h_left+h_up)] (32 x (XKT*32 + 128)) staged in LDS as bf16.
// Two fully-unrolled K loops keep gx / gh accumulators separate (the n-gate
// needs r * gh_n), with zero runtime branching around the WMMAs.
// Wave w owns N-tiles {w, w+8, w+16}  ->  matched (r,z,n) columns, gate math
// is wave-local.
// ---------------------------------------------------------------------------
template<int XKT>   // XKT = in_dim/32: 2 (layer 0) or 4 (layers 1-2)
__global__ __launch_bounds__(256) void diag_scan_kernel(
    const __bf16* __restrict__ xin,   // layer0: emb (H,W,B,64); else h_prev (NDIR,H,W,B,128)
    __bf16* __restrict__ hout,        // (NDIR,H,W,B,128) — this layer's recurrent map
    const __bf16* __restrict__ wxp,   // [NDIR][XKT][24][32][16] bf16
    const __bf16* __restrict__ whp,   // [NDIR][4][24][32][16] bf16
    const float* __restrict__ bx,     // (NDIR,384)
    const float* __restrict__ bh,     // (NDIR,384)
    int t, int cnt)
{
    constexpr int IN_DIM = XKT * 32;
    __shared__ __bf16 A_lds[32][ASTRIDE];
    __shared__ float  Hav[32][HIDN];

    const int d   = blockIdx.x / cnt;
    const int ci  = blockIdx.x % cnt;
    const int i0  = (t > 31) ? (t - 31) : 0;
    const int i   = i0 + ci;
    const int j   = t - i;
    const int tid = threadIdx.x;

    // ---- stage recurrent half: h_avg = 0.5*(h_left + h_up) ----
    for (int idx = tid; idx < 32 * HIDN; idx += 256) {
        int m = idx >> 7, c = idx & 127;
        float hl = 0.f, hu = 0.f;
        if (j > 0) hl = bf2f(hout[((((size_t)d * GRID + i) * GRID + (j - 1)) * BQ + m) * HIDN + c]);
        if (i > 0) hu = bf2f(hout[((((size_t)d * GRID + (i - 1)) * GRID + j) * BQ + m) * HIDN + c]);
        float av = 0.5f * (hl + hu);
        Hav[m][c] = av;
        A_lds[m][IN_DIM + c] = f2bf(av);
    }
    // ---- stage x half (compile-time selected path) ----
    if constexpr (XKT == 2) {                   // layer 0: directional flips of emb
        const int fi = (d & 1) ? (31 - i) : i;
        const int fj = (d & 2) ? (31 - j) : j;
        const __bf16* src = xin + (((size_t)fi * GRID + fj) * BQ) * EMBD;
        for (int idx = tid; idx < 32 * EMBD; idx += 256)
            A_lds[idx >> 6][idx & 63] = src[idx];
    } else {                                    // layers 1-2: previous layer h map
        const __bf16* src = xin + ((((size_t)d * GRID + i) * GRID + j) * BQ) * HIDN;
        for (int idx = tid; idx < 32 * HIDN; idx += 256)
            A_lds[idx >> 7][idx & 127] = src[idx];
    }
    __syncthreads();

    const int w    = tid >> 5;            // wave id = gate column group
    const int lane = tid & 31;
    const int mrow = lane & 15;
    const int kg   = (lane >> 4) << 3;    // 0 or 8 (A-fragment K sub-group)

    v8f accx[2][3], acch[2][3];
    #pragma unroll
    for (int mt = 0; mt < 2; ++mt)
        #pragma unroll
        for (int g = 0; g < 3; ++g) {
            accx[mt][g] = (v8f){0.f,0.f,0.f,0.f,0.f,0.f,0.f,0.f};
            acch[mt][g] = (v8f){0.f,0.f,0.f,0.f,0.f,0.f,0.f,0.f};
        }

    const __bf16* wxd = wxp + (size_t)d * XKT * 24 * 512;
    const __bf16* whd = whp + (size_t)d * 4   * 24 * 512;

    // ---- x-part GEMM: accx += A_x @ Wx ----
    #pragma unroll
    for (int kt = 0; kt < XKT; ++kt) {
        const int kbase = kt * 32;
        v8bf lo0 = *(const v8bf*)&A_lds[mrow][kbase + kg];
        v8bf hi0 = *(const v8bf*)&A_lds[mrow][kbase + 16 + kg];
        v8bf lo1 = *(const v8bf*)&A_lds[16 + mrow][kbase + kg];
        v8bf hi1 = *(const v8bf*)&A_lds[16 + mrow][kbase + 16 + kg];
        v16bf a0 = __builtin_shufflevector(lo0, hi0, 0,1,2,3,4,5,6,7,8,9,10,11,12,13,14,15);
        v16bf a1 = __builtin_shufflevector(lo1, hi1, 0,1,2,3,4,5,6,7,8,9,10,11,12,13,14,15);
        const __bf16* wbase = wxd + (size_t)kt * 24 * 512;
        #pragma unroll
        for (int g = 0; g < 3; ++g) {
            v16bf bf = *(const v16bf*)(wbase + ((size_t)(g * 8 + w) * 32 + lane) * 16);
            accx[0][g] = __builtin_amdgcn_wmma_f32_16x16x32_bf16(false, a0, false, bf, (short)0, accx[0][g], false, false);
            accx[1][g] = __builtin_amdgcn_wmma_f32_16x16x32_bf16(false, a1, false, bf, (short)0, accx[1][g], false, false);
        }
    }
    // ---- h-part GEMM: acch += A_h @ Wh ----
    #pragma unroll
    for (int kt = 0; kt < 4; ++kt) {
        const int kbase = IN_DIM + kt * 32;
        v8bf lo0 = *(const v8bf*)&A_lds[mrow][kbase + kg];
        v8bf hi0 = *(const v8bf*)&A_lds[mrow][kbase + 16 + kg];
        v8bf lo1 = *(const v8bf*)&A_lds[16 + mrow][kbase + kg];
        v8bf hi1 = *(const v8bf*)&A_lds[16 + mrow][kbase + 16 + kg];
        v16bf a0 = __builtin_shufflevector(lo0, hi0, 0,1,2,3,4,5,6,7,8,9,10,11,12,13,14,15);
        v16bf a1 = __builtin_shufflevector(lo1, hi1, 0,1,2,3,4,5,6,7,8,9,10,11,12,13,14,15);
        const __bf16* wbase = whd + (size_t)kt * 24 * 512;
        #pragma unroll
        for (int g = 0; g < 3; ++g) {
            v16bf bf = *(const v16bf*)(wbase + ((size_t)(g * 8 + w) * 32 + lane) * 16);
            acch[0][g] = __builtin_amdgcn_wmma_f32_16x16x32_bf16(false, a0, false, bf, (short)0, acch[0][g], false, false);
            acch[1][g] = __builtin_amdgcn_wmma_f32_16x16x32_bf16(false, a1, false, bf, (short)0, acch[1][g], false, false);
        }
    }

    // ---- GRU gate math (wave-local; column fixed per lane) ----
    const int col = 16 * w + mrow;                 // 0..127 within each gate
    const float bxr = bx[d * 384 + col];
    const float bxz = bx[d * 384 + 128 + col];
    const float bxn = bx[d * 384 + 256 + col];
    const float bhr = bh[d * 384 + col];
    const float bhz = bh[d * 384 + 128 + col];
    const float bhn = bh[d * 384 + 256 + col];
    const int mh = (lane >> 4) << 3;               // 0 or 8 (D-layout M offset)

    #pragma unroll
    for (int mt = 0; mt < 2; ++mt) {
        #pragma unroll
        for (int e = 0; e < 8; ++e) {
            float r = 1.f / (1.f + __expf(-(accx[mt][0][e] + bxr + acch[mt][0][e] + bhr)));
            float z = 1.f / (1.f + __expf(-(accx[mt][1][e] + bxz + acch[mt][1][e] + bhz)));
            float n = tanhf(accx[mt][2][e] + bxn + r * (acch[mt][2][e] + bhn));
            int m = mt * 16 + mh + e;              // batch row
            float hnew = (1.f - z) * n + z * Hav[m][col];
            hout[((((size_t)d * GRID + i) * GRID + j) * BQ + m) * HIDN + col] = f2bf(hnew);
        }
    }
}

// ---------------------------------------------------------------------------
// Pack h_final[:, -1, -1] -> (32 x 512) A-fragment layout [mt][kt][lane][16].
// Row b, column k = d*128+c.  A-frag element map: e<8 -> kt*32+kg+e,
// e>=8 -> kt*32+16+kg+(e-8), kg = (lane>=16)?8:0.
// ---------------------------------------------------------------------------
__global__ void hnpack_kernel(const __bf16* __restrict__ hfin, __bf16* __restrict__ dst) {
    int idx = blockIdx.x * 256 + threadIdx.x;      // 16384 total
    if (idx >= 2 * 16 * 512) return;
    int e  = idx & 15;
    int L  = (idx >> 4) & 31;
    int kt = (idx >> 9) & 15;
    int mt = idx >> 13;
    int kg = (L >= 16) ? 8 : 0;
    int k  = kt * 32 + ((e < 8) ? (kg + e) : (16 + kg + (e - 8)));
    int m  = mt * 16 + (L & 15);
    int dd = k >> 7, c = k & 127;
    dst[idx] = hfin[((((size_t)dd * GRID + 31) * GRID + 31) * BQ + m) * HIDN + c];
}

// ---------------------------------------------------------------------------
// Classifier: out(32x1000) = hn(32x512) @ Wo(512x1000) + bo.  One wave per
// N-tile (63 tiles, last partially masked on store only).
// ---------------------------------------------------------------------------
__global__ __launch_bounds__(32) void final_gemm_kernel(
    const __bf16* __restrict__ ap, const __bf16* __restrict__ wop,
    const float* __restrict__ bo, float* __restrict__ out)
{
    const int nt = blockIdx.x;         // 0..62
    const int lane = threadIdx.x;
    #pragma unroll
    for (int mt = 0; mt < 2; ++mt) {
        v8f acc = (v8f){0.f,0.f,0.f,0.f,0.f,0.f,0.f,0.f};
        #pragma unroll 4
        for (int kt = 0; kt < 16; ++kt) {
            v16bf a = *(const v16bf*)(ap  + (((size_t)mt * 16 + kt) * 32 + lane) * 16);
            v16bf b = *(const v16bf*)(wop + (((size_t)kt * 63 + nt) * 32 + lane) * 16);
            acc = __builtin_amdgcn_wmma_f32_16x16x32_bf16(false, a, false, b, (short)0, acc, false, false);
        }
        const int col = nt * 16 + (lane & 15);
        if (col < NCLSS) {
            const float bias = bo[col];
            #pragma unroll
            for (int e = 0; e < 8; ++e) {
                int m = mt * 16 + ((lane >= 16) ? 8 : 0) + e;
                out[(size_t)m * NCLSS + col] = acc[e] + bias;
            }
        }
    }
}

// ---------------------------------------------------------------------------
extern "C" void kernel_launch(void* const* d_in, const int* in_sizes, int n_in,
                              void* d_out, int out_size, void* d_ws, size_t ws_size,
                              hipStream_t stream) {
    (void)in_sizes; (void)n_in; (void)out_size; (void)ws_size;
    const float* x  = (const float*)d_in[0];
    const float* We = (const float*)d_in[1];
    const float* be = (const float*)d_in[2];
    const float *Wx[3], *Wh[3], *bx[3], *bh[3];
    for (int l = 0; l < 3; ++l) {
        Wx[l] = (const float*)d_in[3 + 4 * l];
        Wh[l] = (const float*)d_in[4 + 4 * l];
        bx[l] = (const float*)d_in[5 + 4 * l];
        bh[l] = (const float*)d_in[6 + 4 * l];
    }
    const float* Wo = (const float*)d_in[15];
    const float* bo = (const float*)d_in[16];
    float* out = (float*)d_out;

    char* ws = (char*)d_ws;
    size_t off = 0;
    auto alloc = [&](size_t bytes) -> char* {
        char* p = ws + off;
        off += (bytes + 255) & ~(size_t)255;
        return p;
    };

    const int indim[3] = {64, 128, 128};
    __bf16* emb = (__bf16*)alloc((size_t)GRID * GRID * BQ * EMBD * 2);
    __bf16 *wxp[3], *whp[3];
    for (int l = 0; l < 3; ++l) {
        wxp[l] = (__bf16*)alloc((size_t)NDIRS * (indim[l] / 32) * 24 * 512 * 2);
        whp[l] = (__bf16*)alloc((size_t)NDIRS * 4 * 24 * 512 * 2);
    }
    __bf16* wop = (__bf16*)alloc((size_t)16 * 63 * 512 * 2);
    __bf16* hA  = (__bf16*)alloc((size_t)NDIRS * GRID * GRID * BQ * HIDN * 2);
    __bf16* hB  = (__bf16*)alloc((size_t)NDIRS * GRID * GRID * BQ * HIDN * 2);
    __bf16* hnp = (__bf16*)alloc((size_t)2 * 16 * 512 * 2);

    // 1) prepack all weights into wave32 B-fragment layout (bf16)
    for (int l = 0; l < 3; ++l) {
        const int K = indim[l];
        for (int d = 0; d < NDIRS; ++d) {
            int tot = (K / 32) * 24 * 512;
            prepack_kernel<<<(tot + 255) / 256, 256, 0, stream>>>(
                Wx[l] + (size_t)d * K * 384, wxp[l] + (size_t)d * (K / 32) * 24 * 512,
                K, 384, 24);
            int tot2 = 4 * 24 * 512;
            prepack_kernel<<<(tot2 + 255) / 256, 256, 0, stream>>>(
                Wh[l] + (size_t)d * 128 * 384, whp[l] + (size_t)d * 4 * 24 * 512,
                128, 384, 24);
        }
    }
    {
        int tot = 16 * 63 * 512;
        prepack_kernel<<<(tot + 255) / 256, 256, 0, stream>>>(Wo, wop, 512, 1000, 63);
    }

    // 2) embedding (bf16)
    embed_kernel<<<8192, 256, 0, stream>>>(x, We, be, emb);

    // 3) three stacked 2-D GRU layers: 63 anti-diagonal wavefront steps each
    __bf16* maps[2] = {hA, hB};
    for (int l = 0; l < 3; ++l) {
        __bf16* hout = maps[l & 1];                       // l0->hA, l1->hB, l2->hA
        const __bf16* xin = (l == 0) ? emb : maps[(l - 1) & 1];
        for (int t = 0; t < 63; ++t) {
            int i0 = (t > 31) ? (t - 31) : 0;
            int i1 = (t < 31) ? t : 31;
            int cnt = i1 - i0 + 1;
            if (l == 0)
                diag_scan_kernel<2><<<cnt * NDIRS, 256, 0, stream>>>(
                    xin, hout, wxp[l], whp[l], bx[l], bh[l], t, cnt);
            else
                diag_scan_kernel<4><<<cnt * NDIRS, 256, 0, stream>>>(
                    xin, hout, wxp[l], whp[l], bx[l], bh[l], t, cnt);
        }
    }

    // 4) gather h[:, -1, -1] into WMMA A layout, then classifier GEMM
    hnpack_kernel<<<64, 256, 0, stream>>>(maps[0], hnp);
    final_gemm_kernel<<<63, 32, 0, stream>>>(hnp, wop, bo, out);
}
